// AllOutputsGRU_47837345743386
// MI455X (gfx1250) — matrix operand order repaired and checked by reading, not compiled, not run on hardware
//
#include <hip/hip_runtime.h>
#include <cstdint>
#include <cstddef>

// ---------------------------------------------------------------------------
// AllOutputsGRU for MI455X (gfx1250, wave32, WMMA)
//   T=1024 B=64 F=256 H=512 L=2 ; both layers consume x independently.
//   Per step fused GEMM: A = [x_t | h] (64 x 768 bf16, LDS)
//                        Bw = [W_ih | W_hh] (1536 x 768 bf16, L2-resident)
//   v_wmma_f32_16x16x32_bf16, f32 accumulate, gates fused in-register.
//   Each wave owns one N-tile (jcol) and TWO M-tiles -> B fragments loaded
//   once and reused across both M-tiles (6 WMMA per 3 B loads).
//   K-loop split into x-phase (kc<8 -> aNx) and h-phase (kc>=8 -> aNh) so no
//   conditional accumulators (round-1 asm showed branchy select + v_nops).
// ---------------------------------------------------------------------------

typedef __bf16 bf16;
typedef __attribute__((ext_vector_type(16))) __bf16 v16bf;
typedef __attribute__((ext_vector_type(8)))  __bf16 v8bf;
typedef __attribute__((ext_vector_type(4)))  __bf16 v4bf;
typedef __attribute__((ext_vector_type(8)))  float  v8f;

#define T_ 1024
#define B_ 64
#define F_ 256
#define H_ 512
#define L_ 2
#define G3_ 1536                 // 3*H
#define K_ 768                   // F + H
#define NB_ 8                    // blocks per layer (split over H columns)
#define NCOL_ (H_ / NB_)         // 64 hidden columns per block
#define APITCH 776               // LDS row pitch in bf16 (768 + 8 pad)
#define THREADS 256              // 8 waves of 32

union Frag { v16bf v; v8bf h[2]; };

static __device__ __forceinline__ v8f wmma_bf16(v16bf a, v16bf b, v8f c) {
  // (neg_a, A, neg_b, B, c_mod, C, reuse_a, reuse_b)
  return __builtin_amdgcn_wmma_f32_16x16x32_bf16(false, a, false, b, (short)0, c,
                                                 false, false);
}
static __device__ __forceinline__ float sigm_(float x) {
  return 1.0f / (1.0f + __expf(-x));
}
static __device__ __forceinline__ float tanh_(float x) {
  float e = __expf(2.0f * x);
  return (e - 1.0f) / (e + 1.0f);
}
static __device__ __forceinline__ Frag ldsA(const bf16* arow, int kb) {
  Frag a;
  a.h[0] = *(const v8bf*)(arow + kb);        // K = kb + 8*g2 .. +7
  a.h[1] = *(const v8bf*)(arow + kb + 16);   // K = kb + 16 + 8*g2 .. +7
  return a;
}
static __device__ __forceinline__ Frag ldB(const bf16* w, int kb) {
  Frag b;
  b.h[0] = *(const v8bf*)(w + kb);           // K = kb + 8*g2 .. +7
  b.h[1] = *(const v8bf*)(w + kb + 16);      // K = kb + 16 + 8*g2 .. +7
  return b;
}

// --- prep: Wc[l][g][k] = bf16( k<F ? W_ih[l][g][k] : W_hh[l][g][k-F] ) ------
__global__ void gru_prep_weights(const float* __restrict__ Wih,
                                 const float* __restrict__ Whh,
                                 bf16* __restrict__ Wc) {
  const size_t total = (size_t)L_ * G3_ * K_;
  for (size_t i = (size_t)blockIdx.x * blockDim.x + threadIdx.x; i < total;
       i += (size_t)gridDim.x * blockDim.x) {
    size_t l = i / ((size_t)G3_ * K_);
    size_t r = i % ((size_t)G3_ * K_);
    size_t g = r / K_, k = r % K_;
    float w = (k < F_) ? Wih[(l * G3_ + g) * F_ + k]
                       : Whh[(l * G3_ + g) * H_ + (k - F_)];
    Wc[i] = (bf16)w;
  }
}

__global__ void gru_zero_f32(float* __restrict__ p, size_t n) {
  for (size_t i = (size_t)blockIdx.x * blockDim.x + threadIdx.x; i < n;
       i += (size_t)gridDim.x * blockDim.x) p[i] = 0.0f;
}

// --- gate epilogue for one 16x16 tile position ------------------------------
static __device__ __forceinline__ void gru_epilogue(
    v8f aR, v8f aZ, v8f aNx, v8f aNh, int mi, int jcol, int g2,
    float bRc, float bZc, float bNx, float bNh,
    const float* __restrict__ hread, float* __restrict__ hwrite,
    float* __restrict__ out, int t) {
  #pragma unroll
  for (int v = 0; v < 8; ++v) {                   // D layout: m = v + 8*g2
    int b = mi * 16 + v + 8 * g2;
    float r = sigm_(aR[v] + bRc);
    float z = sigm_(aZ[v] + bZc);
    float n = tanh_(aNx[v] + bNx + r * (aNh[v] + bNh));
    float hp = hread[b * H_ + jcol];
    float hn = (1.0f - z) * n + z * hp;
    hwrite[b * H_ + jcol] = hn;
    atomicAdd(out + ((size_t)t * B_ + b) * H_ + jcol, 0.5f * hn);
  }
}

// --- persistent recurrent kernel: grid = (NB_, L_), 256 threads -------------
__global__ void __launch_bounds__(THREADS)
gru_step_kernel(const float* __restrict__ x, const float* __restrict__ bih_all,
                const float* __restrict__ bhh_all, const bf16* __restrict__ Wc_all,
                float* __restrict__ hbuf_all, unsigned* __restrict__ bars,
                float* __restrict__ out) {
  extern __shared__ __align__(16) char smem_raw[];
  bf16* Abuf = (bf16*)smem_raw;                    // 64 x APITCH bf16

  const int layer = blockIdx.y;
  const int c0 = blockIdx.x * NCOL_;               // first hidden col owned
  const int tid = threadIdx.x;
  const int lane = tid & 31;
  const int wave = tid >> 5;
  const int g2 = lane >> 4;                        // half-wave id (K phase)
  const int nl = lane & 15;

  // wave -> one N-tile (cj) and two M-tiles (mi0, mi0+1): shared B fragments
  const int mi0 = (wave & 1) * 2;
  const int cj = wave >> 1;

  const bf16* Wc = Wc_all + (size_t)layer * G3_ * K_;
  const float* bih = bih_all + layer * G3_;
  const float* bhh = bhh_all + layer * G3_;
  float* hb0 = hbuf_all + (size_t)layer * 2 * B_ * H_;
  unsigned* bar = bars + layer;

  const int jcol = c0 + cj * 16 + nl;              // hidden column 0..511
  const bf16* arow0 = Abuf + (mi0 * 16 + nl) * APITCH + 8 * g2;
  const bf16* arow1 = arow0 + 16 * APITCH;
  const bf16* wR = Wc + (size_t)jcol * K_ + 8 * g2;
  const bf16* wZ = Wc + (size_t)(H_ + jcol) * K_ + 8 * g2;
  const bf16* wN = Wc + (size_t)(2 * H_ + jcol) * K_ + 8 * g2;
  const float bRc = bih[jcol] + bhh[jcol];
  const float bZc = bih[H_ + jcol] + bhh[H_ + jcol];
  const float bNx = bih[2 * H_ + jcol];
  const float bNh = bhh[2 * H_ + jcol];

  for (int t = 0; t < T_; ++t) {
    const float* hread = hb0 + (t & 1) * (B_ * H_);
    float* hwrite = hb0 + ((t + 1) & 1) * (B_ * H_);

    // ---- stage A = [x_t | h_{t-1}] into LDS as bf16 ----
    const float* xt = x + (size_t)t * B_ * F_;
    for (int idx = tid; idx < (B_ * F_) / 4; idx += THREADS) {
      int b = idx >> 6, f = (idx & 63) << 2;
      float4 v = *(const float4*)(xt + b * F_ + f);
      v4bf o; o.x = (bf16)v.x; o.y = (bf16)v.y; o.z = (bf16)v.z; o.w = (bf16)v.w;
      *(v4bf*)(Abuf + b * APITCH + f) = o;
    }
    for (int idx = tid; idx < (B_ * H_) / 4; idx += THREADS) {
      int b = idx >> 7, c = (idx & 127) << 2;
      float4 v = *(const float4*)(hread + b * H_ + c);
      v4bf o; o.x = (bf16)v.x; o.y = (bf16)v.y; o.z = (bf16)v.z; o.w = (bf16)v.w;
      *(v4bf*)(Abuf + b * APITCH + F_ + c) = o;
    }
    __syncthreads();

    v8f aR0 = {}, aZ0 = {}, aNx0 = {}, aNh0 = {};
    v8f aR1 = {}, aZ1 = {}, aNx1 = {}, aNh1 = {};

    // ---- x-phase: K = 0..255 (n-gate accumulates into aNx) ----
    #pragma unroll 4
    for (int kc = 0; kc < F_ / 32; ++kc) {
      int kb = kc * 32;
      Frag a0 = ldsA(arow0, kb), a1 = ldsA(arow1, kb);
      Frag bR = ldB(wR, kb), bZ = ldB(wZ, kb), bN = ldB(wN, kb);
      aR0 = wmma_bf16(a0.v, bR.v, aR0);  aR1 = wmma_bf16(a1.v, bR.v, aR1);
      aZ0 = wmma_bf16(a0.v, bZ.v, aZ0);  aZ1 = wmma_bf16(a1.v, bZ.v, aZ1);
      aNx0 = wmma_bf16(a0.v, bN.v, aNx0); aNx1 = wmma_bf16(a1.v, bN.v, aNx1);
    }
    // ---- h-phase: K = 256..767 (n-gate accumulates into aNh) ----
    #pragma unroll 4
    for (int kc = F_ / 32; kc < K_ / 32; ++kc) {
      int kb = kc * 32;
      Frag a0 = ldsA(arow0, kb), a1 = ldsA(arow1, kb);
      Frag bR = ldB(wR, kb), bZ = ldB(wZ, kb), bN = ldB(wN, kb);
      aR0 = wmma_bf16(a0.v, bR.v, aR0);  aR1 = wmma_bf16(a1.v, bR.v, aR1);
      aZ0 = wmma_bf16(a0.v, bZ.v, aZ0);  aZ1 = wmma_bf16(a1.v, bZ.v, aZ1);
      aNh0 = wmma_bf16(a0.v, bN.v, aNh0); aNh1 = wmma_bf16(a1.v, bN.v, aNh1);
    }

    gru_epilogue(aR0, aZ0, aNx0, aNh0, mi0,     jcol, g2, bRc, bZc, bNx, bNh,
                 hread, hwrite, out, t);
    gru_epilogue(aR1, aZ1, aNx1, aNh1, mi0 + 1, jcol, g2, bRc, bZc, bNx, bNh,
                 hread, hwrite, out, t);

    // ---- per-layer grid barrier (release/acquire on monotone counter) ----
    __syncthreads();
    if (tid == 0) {
      __hip_atomic_fetch_add(bar, 1u, __ATOMIC_RELEASE, __HIP_MEMORY_SCOPE_AGENT);
      const unsigned target = (unsigned)(NB_ * (t + 1));
      while (__hip_atomic_load(bar, __ATOMIC_ACQUIRE, __HIP_MEMORY_SCOPE_AGENT)
             < target) {
        __builtin_amdgcn_s_sleep(1);
      }
    }
    __syncthreads();
  }
}

extern "C" void kernel_launch(void* const* d_in, const int* in_sizes, int n_in,
                              void* d_out, int out_size, void* d_ws, size_t ws_size,
                              hipStream_t stream) {
  const float* x   = (const float*)d_in[0];   // (T,B,F)
  const float* Wih = (const float*)d_in[1];   // (L,3H,F)
  const float* Whh = (const float*)d_in[2];   // (L,3H,H)
  const float* bih = (const float*)d_in[3];   // (L,3H)
  const float* bhh = (const float*)d_in[4];   // (L,3H)
  float* out = (float*)d_out;                 // (T,B,H)
  char* ws = (char*)d_ws;

  const size_t WC_BYTES = (size_t)L_ * G3_ * K_ * sizeof(bf16);  // 4,718,592
  const size_t HB_ELEMS = (size_t)L_ * 2 * B_ * H_;              // 131072 f32
  bf16* Wc = (bf16*)ws;
  float* hbuf = (float*)(ws + WC_BYTES);
  unsigned* bars = (unsigned*)(ws + WC_BYTES + HB_ELEMS * sizeof(float));

  gru_prep_weights<<<1024, 256, 0, stream>>>(Wih, Whh, Wc);
  gru_zero_f32<<<2048, 256, 0, stream>>>(out, (size_t)out_size);
  gru_zero_f32<<<64, 256, 0, stream>>>(hbuf, HB_ELEMS + 16);  // +bars region

  dim3 grid(NB_, L_);
  size_t smem = (size_t)B_ * APITCH * sizeof(bf16);            // ~99 KB
  gru_step_kernel<<<grid, THREADS, smem, stream>>>(x, bih, bhh, Wc, hbuf, bars,
                                                   out);
}
